// temporalShiftedAttentionSignal_37185826848898
// MI455X (gfx1250) — matrix operand order repaired and checked by reading, not compiled, not run on hardware
//
#include <hip/hip_runtime.h>

typedef unsigned short u16;
typedef unsigned long long u64;
typedef __attribute__((ext_vector_type(16))) __bf16 v16bf;
typedef __attribute__((ext_vector_type(8)))  float  v8f;
typedef __attribute__((ext_vector_type(4)))  unsigned int v4u;

#define WMMA_BF16(a, b, c) \
  __builtin_amdgcn_wmma_f32_16x16x32_bf16(false, (a), false, (b), (short)0, (c), false, false)

#define L_TOK 3072
#define C_DIM 1024
#define NHEAD 16
#define SEQ_LEN 1024

__device__ __forceinline__ u16 f2bf(float f) {
  unsigned u = __float_as_uint(f);
  return (u16)((u + 0x7fffu + ((u >> 16) & 1u)) >> 16);  // RNE truncate
}
__device__ __forceinline__ float bf2f(u16 h) {
  return __uint_as_float((unsigned)h << 16);
}

union V16BF { v16bf v; struct { v4u x, y; } p; };

// A operand (16x32 bf16, M x K). Lanes 0-15: K[0..7] in V0-3, K[16..23] in V4-7.
// Lanes 16-31: K[8..15], K[24..31]. Source: row-major rows over K in LDS.
__device__ __forceinline__ v16bf frag_a(const u16* base, int row, int stride, int kcol, int lane) {
  const u16* p = base + row * stride + kcol + ((lane >> 4) << 3);
  V16BF t;
  t.p.x = *(const v4u*)(p);
  t.p.y = *(const v4u*)(p + 16);
  return t.v;
}
// B operand (32x16 bf16, K x N), B[k][n] = src_row_n[k]. Lanes 0-15: K[0..7],
// K[16..23]; lanes 16-31: K[8..15], K[24..31] (same as A).
__device__ __forceinline__ v16bf frag_b(const u16* base, int row, int stride, int kcol, int lane) {
  const u16* p = base + row * stride + kcol + ((lane >> 4) << 3);
  V16BF t;
  t.p.x = *(const v4u*)(p);
  t.p.y = *(const v4u*)(p + 16);
  return t.v;
}

// ---------------- split fp32 -> bf16 hi/lo ----------------
__global__ __launch_bounds__(256) void split_kernel(const float4* __restrict__ src,
                                                    u64* __restrict__ hi,
                                                    u64* __restrict__ lo, int n4) {
  int i = blockIdx.x * 256 + threadIdx.x;
  if (i >= n4) return;
  float4 v = src[i];
  u16 h0 = f2bf(v.x), h1 = f2bf(v.y), h2 = f2bf(v.z), h3 = f2bf(v.w);
  u16 l0 = f2bf(v.x - bf2f(h0)), l1 = f2bf(v.y - bf2f(h1));
  u16 l2 = f2bf(v.z - bf2f(h2)), l3 = f2bf(v.w - bf2f(h3));
  hi[i] = (u64)h0 | ((u64)h1 << 16) | ((u64)h2 << 32) | ((u64)h3 << 48);
  lo[i] = (u64)l0 | ((u64)l1 << 16) | ((u64)l2 << 32) | ((u64)l3 << 48);
}

__global__ __launch_bounds__(256) void zero_kernel(float4* __restrict__ p, int n4) {
  int i = blockIdx.x * 256 + threadIdx.x;
  if (i < n4) p[i] = make_float4(0.f, 0.f, 0.f, 0.f);
}

// ---------------- QKV GEMM (q,k only) + bias + RoPE ----------------
// Block: 64 rows (tokens) x 64 cols (= one head of q or of k). 4 waves,
// wave wv owns the 16-row strip, 4 accumulator tiles across N.
__global__ __launch_bounds__(128) void qkv_rope_kernel(
    const u16* __restrict__ x_hi, const u16* __restrict__ x_lo,
    const u16* __restrict__ w_hi, const u16* __restrict__ w_lo,
    const float* __restrict__ qkv_b,
    float* __restrict__ qbuf, float* __restrict__ kbuf) {
  __shared__ u16 sAh[64 * 32], sAl[64 * 32], sBh[64 * 32], sBl[64 * 32];

  const int n0 = blockIdx.x * 64;   // 0..2047 (q cols 0..1023, k cols 1024..2047)
  const int l0 = blockIdx.y * 64;   // token base
  const int t = threadIdx.x;
  const int lrow = t >> 1;          // 0..63
  const int lseg = (t & 1) << 4;    // 0 or 16 (u16 elems)
  const int lane = t & 31;
  const int wv = t >> 5;
  const int lane15 = lane & 15;

  v8f acc[4] = {};

  for (int kk = 0; kk < C_DIM; kk += 32) {
    {
      size_t ga = (size_t)(l0 + lrow) * C_DIM + kk + lseg;
      size_t gb = (size_t)(n0 + lrow) * C_DIM + kk + lseg;
      int ls = lrow * 32 + lseg;
      *(v4u*)&sAh[ls]     = *(const v4u*)&x_hi[ga];
      *(v4u*)&sAh[ls + 8] = *(const v4u*)&x_hi[ga + 8];
      *(v4u*)&sAl[ls]     = *(const v4u*)&x_lo[ga];
      *(v4u*)&sAl[ls + 8] = *(const v4u*)&x_lo[ga + 8];
      *(v4u*)&sBh[ls]     = *(const v4u*)&w_hi[gb];
      *(v4u*)&sBh[ls + 8] = *(const v4u*)&w_hi[gb + 8];
      *(v4u*)&sBl[ls]     = *(const v4u*)&w_lo[gb];
      *(v4u*)&sBl[ls + 8] = *(const v4u*)&w_lo[gb + 8];
    }
    __syncthreads();
    v16bf ah = frag_a(sAh, wv * 16 + lane15, 32, 0, lane);
    v16bf al = frag_a(sAl, wv * 16 + lane15, 32, 0, lane);
#pragma unroll
    for (int nt = 0; nt < 4; ++nt) {
      v16bf bh = frag_b(sBh, nt * 16 + lane15, 32, 0, lane);
      v16bf bl = frag_b(sBl, nt * 16 + lane15, 32, 0, lane);
      acc[nt] = WMMA_BF16(ah, bh, acc[nt]);   // hi*hi
      acc[nt] = WMMA_BF16(ah, bl, acc[nt]);   // hi*lo
      acc[nt] = WMMA_BF16(al, bh, acc[nt]);   // lo*hi
    }
    __syncthreads();
  }

  // Epilogue: bias + rotary. Tile pair (p, p+2) holds head-dims d and d+32
  // in the same lane/VGPR (d = p*16 + lane15).
  const int hb = lane >> 4;
  float biasv[4];
#pragma unroll
  for (int nt = 0; nt < 4; ++nt) biasv[nt] = qkv_b[n0 + nt * 16 + lane15];
  float* dst = (n0 < 1024) ? qbuf : kbuf;
  const int colb = n0 & 1023;  // head*64 within the q or k buffer

#pragma unroll
  for (int j = 0; j < 8; ++j) {
    int m = j + (hb << 3);
    int l = l0 + wv * 16 + m;
    size_t rb = (size_t)l * C_DIM + colb;
#pragma unroll
    for (int p = 0; p < 2; ++p) {
      float v1 = acc[p][j] + biasv[p];
      float v2 = acc[p + 2][j] + biasv[p + 2];
      int d = p * 16 + lane15;                      // freq index in [0,32)
      float invf = exp2f((float)d * -0.41524101186092030f);  // 10000^(-d/32)
      float ang = (float)l * invf;
      float c = cosf(ang), s = sinf(ang);
      dst[rb + d]      = v1 * c - v2 * s;
      dst[rb + 32 + d] = v1 * s + v2 * c;
    }
  }
}

// ---------------- windowed attention + softmax ----------------
// Block = (window, seq, head). W in {32,64}; S = (Q_w K_w^T)/8; row softmax.
__global__ __launch_bounds__(128) void win_attn_kernel(
    const float* __restrict__ qbuf, const float* __restrict__ kbuf,
    float* __restrict__ out) {
  __shared__ u16 sQh[64 * 64], sQl[64 * 64], sKh[64 * 64], sKl[64 * 64];

  const int w = blockIdx.x;   // 0..16
  const int s = blockIdx.y;   // 0..2
  const int h = blockIdx.z;   // 0..15
  const int W = (w == 0 || w == 16) ? 32 : 64;
  const int pos0 = (w == 0) ? 0 : (w * 64 - 32);
  const int l0 = s * SEQ_LEN + pos0;
  const int t = threadIdx.x;
  const int hofs = h * 64;

  // Stage q/k window panels into LDS as bf16 hi/lo.
  for (int cid = t; cid < W * 16; cid += 128) {
    int r = cid >> 4;
    int c = (cid & 15) << 2;
    size_t g = (size_t)(l0 + r) * C_DIM + hofs + c;
    float4 qv = *(const float4*)&qbuf[g];
    float4 kv = *(const float4*)&kbuf[g];
    {
      u16 h0 = f2bf(qv.x), h1 = f2bf(qv.y), h2 = f2bf(qv.z), h3 = f2bf(qv.w);
      u16 e0 = f2bf(qv.x - bf2f(h0)), e1 = f2bf(qv.y - bf2f(h1));
      u16 e2 = f2bf(qv.z - bf2f(h2)), e3 = f2bf(qv.w - bf2f(h3));
      *(u64*)&sQh[r * 64 + c] = (u64)h0 | ((u64)h1 << 16) | ((u64)h2 << 32) | ((u64)h3 << 48);
      *(u64*)&sQl[r * 64 + c] = (u64)e0 | ((u64)e1 << 16) | ((u64)e2 << 32) | ((u64)e3 << 48);
    }
    {
      u16 h0 = f2bf(kv.x), h1 = f2bf(kv.y), h2 = f2bf(kv.z), h3 = f2bf(kv.w);
      u16 e0 = f2bf(kv.x - bf2f(h0)), e1 = f2bf(kv.y - bf2f(h1));
      u16 e2 = f2bf(kv.z - bf2f(h2)), e3 = f2bf(kv.w - bf2f(h3));
      *(u64*)&sKh[r * 64 + c] = (u64)h0 | ((u64)h1 << 16) | ((u64)h2 << 32) | ((u64)h3 << 48);
      *(u64*)&sKl[r * 64 + c] = (u64)e0 | ((u64)e1 << 16) | ((u64)e2 << 32) | ((u64)e3 << 48);
    }
  }
  __syncthreads();

  const int lane = t & 31;
  const int wv = t >> 5;
  const int lane15 = lane & 15;
  const int NT = W >> 4;        // 2 or 4 N-tiles
  if (wv >= NT) return;         // wave-uniform; EXEC stays all-ones for WMMA

  v8f acc[4] = {};
#pragma unroll
  for (int kk = 0; kk < 64; kk += 32) {
    v16bf ah = frag_a(sQh, wv * 16 + lane15, 64, kk, lane);
    v16bf al = frag_a(sQl, wv * 16 + lane15, 64, kk, lane);
#pragma unroll
    for (int nt = 0; nt < 4; ++nt) {
      if (nt < NT) {
        v16bf bh = frag_b(sKh, nt * 16 + lane15, 64, kk, lane);
        v16bf bl = frag_b(sKl, nt * 16 + lane15, 64, kk, lane);
        acc[nt] = WMMA_BF16(ah, bh, acc[nt]);
        acc[nt] = WMMA_BF16(ah, bl, acc[nt]);
        acc[nt] = WMMA_BF16(al, bh, acc[nt]);
      }
    }
  }

  // scale, softmax per row (row m: lanes 0-15 hold VGPR j = m, lanes 16-31 m-8),
  // then scatter the W x W block into the zero-filled output.
  const int hb = lane >> 4;
#pragma unroll
  for (int j = 0; j < 8; ++j) {
    float rmax = -3.0e38f;
#pragma unroll
    for (int nt = 0; nt < 4; ++nt)
      if (nt < NT) { acc[nt][j] *= 0.125f; rmax = fmaxf(rmax, acc[nt][j]); }
#pragma unroll
    for (int md = 1; md < 16; md <<= 1) rmax = fmaxf(rmax, __shfl_xor(rmax, md, 32));
    float rsum = 0.f;
#pragma unroll
    for (int nt = 0; nt < 4; ++nt)
      if (nt < NT) { float e = expf(acc[nt][j] - rmax); acc[nt][j] = e; rsum += e; }
#pragma unroll
    for (int md = 1; md < 16; md <<= 1) rsum += __shfl_xor(rsum, md, 32);
    float rinv = 1.0f / rsum;
    int row = wv * 16 + j + (hb << 3);
    size_t base = ((size_t)h * L_TOK + (l0 + row)) * L_TOK + l0;
#pragma unroll
    for (int nt = 0; nt < 4; ++nt)
      if (nt < NT) out[base + nt * 16 + lane15] = acc[nt][j] * rinv;
  }
}

extern "C" void kernel_launch(void* const* d_in, const int* in_sizes, int n_in,
                              void* d_out, int out_size, void* d_ws, size_t ws_size,
                              hipStream_t stream) {
  const float* x     = (const float*)d_in[0];
  const float* qkv_w = (const float*)d_in[1];
  const float* qkv_b = (const float*)d_in[2];
  float* out = (float*)d_out;
  char* ws = (char*)d_ws;

  // workspace layout (bytes)
  u16* x_hi = (u16*)(ws + 0);             // 3072*1024*2 = 6291456
  u16* x_lo = (u16*)(ws + 6291456);
  u16* w_hi = (u16*)(ws + 12582912);      // 2048*1024*2 = 4194304
  u16* w_lo = (u16*)(ws + 16777216);
  float* qbuf = (float*)(ws + 20971520);  // 3072*1024*4 = 12582912
  float* kbuf = (float*)(ws + 33554432);  // total 46137344 bytes

  // 1) split x and first 2048 rows of qkv_w (q,k weights) into bf16 hi/lo
  split_kernel<<<3072, 256, 0, stream>>>((const float4*)x, (u64*)x_hi, (u64*)x_lo, 786432);
  split_kernel<<<2048, 256, 0, stream>>>((const float4*)qkv_w, (u64*)w_hi, (u64*)w_lo, 524288);

  // 2) zero-fill the 16x3072x3072 output (masked entries softmax to exactly 0)
  zero_kernel<<<147456, 256, 0, stream>>>((float4*)out, 37748736);

  // 3) QKV GEMM (q,k) + bias + RoPE -> qbuf/kbuf
  qkv_rope_kernel<<<dim3(32, 48), 128, 0, stream>>>(x_hi, x_lo, w_hi, w_lo, qkv_b, qbuf, kbuf);

  // 4) windowed attention blocks + softmax (17 windows x 3 seqs x 16 heads)
  win_attn_kernel<<<dim3(17, 3, 16), 128, 0, stream>>>(qbuf, kbuf, out);
}
